// MultiHeadAttention_50165218017961
// MI455X (gfx1250) — matrix operand order repaired and checked by reading, not compiled, not run on hardware
//
#include <hip/hip_runtime.h>
#include <hip/hip_bf16.h>

// ---------------------------------------------------------------------------
// MHA for MI455X (gfx1250): f16 WMMA (v_wmma_f32_16x16x32_f16) everywhere,
// flash-attention streaming (no 1 GB score matrix), register double-buffered
// GEMM mainloops so vmem overlaps the WMMA pipe.
// B=4, S=2048, D=1024, H=16, HD=64.
// ---------------------------------------------------------------------------

typedef __attribute__((ext_vector_type(16))) _Float16 v16h;
typedef __attribute__((ext_vector_type(8)))  _Float16 v8h;
typedef __attribute__((ext_vector_type(4)))  _Float16 v4h;
typedef __attribute__((ext_vector_type(8)))  float    v8f;
typedef __attribute__((ext_vector_type(4)))  float    v4f;

#define BATCH 4
#define SEQ   2048
#define DMODEL 1024
#define NHEAD 16
#define HDIM  64
#define BH    (BATCH * NHEAD)     // 64
#define MROWS (BATCH * SEQ)       // 8192

__device__ __forceinline__ v16h mk16(v8h lo, v8h hi) {
  v16h r;
#pragma unroll
  for (int i = 0; i < 8; ++i) { r[i] = lo[i]; r[i + 8] = hi[i]; }
  return r;
}

// A operand (16x32 f16): lane (l&15)=row, kb=(l>>4)*8; elems 0..7 = K kb..kb+7,
// elems 8..15 = K 16+kb..16+kb+7.  p points at row_ptr + kb.
__device__ __forceinline__ v16h load_a_tile(const _Float16* p) {
  return mk16(*(const v8h*)p, *(const v8h*)(p + 16));
}
// B operand (32x16 f16): lane (l&15)=col N, kb=(l>>4)*8; elems 0..7 = K kb..kb+7,
// elems 8..15 = K 16+kb..16+kb+7.  p points at col_ptr + kb (K-major source).
__device__ __forceinline__ v16h load_b_tile(const _Float16* p) {
  return mk16(*(const v8h*)p, *(const v8h*)(p + 16));
}

__device__ __forceinline__ v8f wmma_f16(v16h a, v16h b, v8f c) {
  return __builtin_amdgcn_wmma_f32_16x16x32_f16(false, a, false, b, (short)0, c,
                                                false, false);
}

__device__ __forceinline__ float rmax16(float v) {
#pragma unroll
  for (int m = 1; m < 16; m <<= 1) v = fmaxf(v, __shfl_xor(v, m, 32));
  return v;
}
__device__ __forceinline__ float rsum16(float v) {
#pragma unroll
  for (int m = 1; m < 16; m <<= 1) v += __shfl_xor(v, m, 32);
  return v;
}

// ----------------------------- conversion ----------------------------------
// 4 elems / thread: global_load_b128 + global_store_b64
__global__ __launch_bounds__(256) void cvt_h4(const float* __restrict__ src,
                                              _Float16* __restrict__ dst) {
  int i = blockIdx.x * 256 + threadIdx.x;
  v4f x = ((const v4f*)src)[i];
  v4h y;
#pragma unroll
  for (int j = 0; j < 4; ++j) y[j] = (_Float16)x[j];
  ((v4h*)dst)[i] = y;
}

// WT[n*D + k] = W[k*D + n], LDS-tiled 32x32 so both sides stay coalesced.
__global__ __launch_bounds__(256) void cvt_hT(const float* __restrict__ src,
                                              _Float16* __restrict__ dst) {
  __shared__ _Float16 tile[32][33];
  const int bx = blockIdx.x & 31, by = blockIdx.x >> 5;   // 32x32 grid of tiles
  const int tx = threadIdx.x & 31, ty = threadIdx.x >> 5; // 8 rows / pass
  const int r0 = by * 32, c0 = bx * 32;                   // k-range, n-range
#pragma unroll
  for (int j = 0; j < 32; j += 8)
    tile[ty + j][tx] = (_Float16)src[(size_t)(r0 + ty + j) * DMODEL + c0 + tx];
  __syncthreads();
#pragma unroll
  for (int j = 0; j < 32; j += 8)
    dst[(size_t)(c0 + ty + j) * DMODEL + r0 + tx] = tile[tx][ty + j];
}

// ------------------------------ WMMA GEMM ----------------------------------
// C[M=8192, N=1024] = A[8192,1024] (f16 row-major) x Bt[N,K] (f16, K-major).
// One wave -> 32x64 output tile (8 WMMAs / K-step), register double-buffered.
// mode 0: f32 row-major (final projection -> d_out)
// mode 1: f16 split-head  [BH, S, HD]   (Q, K)
// mode 2: f16 split-head transposed [BH, HD, S]  (V)
__global__ __launch_bounds__(128) void gemm_wmma(const _Float16* __restrict__ A,
                                                 const _Float16* __restrict__ Bt,
                                                 void* __restrict__ Cout,
                                                 int mode) {
  const int lane = threadIdx.x & 31;
  const int w    = threadIdx.x >> 5;
  const int wf   = blockIdx.x * 4 + w;     // 4096 waves
  const int mt   = wf >> 4;                // 256 M tiles of 32
  const int nt   = wf & 15;                // 16 N tiles of 64
  const int m0   = mt * 32;
  const int n0   = nt * 64;
  const int r    = lane & 15;
  const int hi   = lane >> 4;

  v8f acc[2][4] = {};
  const _Float16* arow0 = A + (size_t)(m0 + r) * DMODEL + hi * 8;
  const _Float16* arow1 = arow0 + (size_t)16 * DMODEL;
  const _Float16* bcol  = Bt + (size_t)(n0 + r) * DMODEL + hi * 8;

  // prologue: K-step 0 operands
  v16h a0 = load_a_tile(arow0);
  v16h a1 = load_a_tile(arow1);
  v16h b0 = load_b_tile(bcol);
  v16h b1 = load_b_tile(bcol + 16 * DMODEL);
  v16h b2 = load_b_tile(bcol + 32 * DMODEL);
  v16h b3 = load_b_tile(bcol + 48 * DMODEL);

  for (int k = 0; k < DMODEL; k += 32) {
    const int kn = (k + 32) & (DMODEL - 1);   // wraps to 0 on last iter (harmless)
    // issue next K-step loads before consuming current operands
    v16h na0 = load_a_tile(arow0 + kn);
    v16h na1 = load_a_tile(arow1 + kn);
    v16h nb0 = load_b_tile(bcol + kn);
    v16h nb1 = load_b_tile(bcol + kn + 16 * DMODEL);
    v16h nb2 = load_b_tile(bcol + kn + 32 * DMODEL);
    v16h nb3 = load_b_tile(bcol + kn + 48 * DMODEL);

    acc[0][0] = wmma_f16(a0, b0, acc[0][0]);
    acc[1][0] = wmma_f16(a1, b0, acc[1][0]);
    acc[0][1] = wmma_f16(a0, b1, acc[0][1]);
    acc[1][1] = wmma_f16(a1, b1, acc[1][1]);
    acc[0][2] = wmma_f16(a0, b2, acc[0][2]);
    acc[1][2] = wmma_f16(a1, b2, acc[1][2]);
    acc[0][3] = wmma_f16(a0, b3, acc[0][3]);
    acc[1][3] = wmma_f16(a1, b3, acc[1][3]);

    a0 = na0; a1 = na1; b0 = nb0; b1 = nb1; b2 = nb2; b3 = nb3;
  }

  if (mode == 0) {
    float* C = (float*)Cout;
#pragma unroll
    for (int ai = 0; ai < 2; ++ai)
#pragma unroll
      for (int t = 0; t < 4; ++t)
#pragma unroll
        for (int i = 0; i < 8; ++i)
          C[(size_t)(m0 + ai * 16 + 8 * hi + i) * DMODEL + n0 + t * 16 + r] =
              acc[ai][t][i];
  } else {
    _Float16* C = (_Float16*)Cout;
#pragma unroll
    for (int ai = 0; ai < 2; ++ai)
#pragma unroll
      for (int t = 0; t < 4; ++t)
#pragma unroll
        for (int i = 0; i < 8; ++i) {
          int row = m0 + ai * 16 + 8 * hi + i;   // b*S + s
          int col = n0 + t * 16 + r;             // h*HD + hd
          int b = row >> 11, s = row & 2047;
          int h = col >> 6,  hd = col & 63;
          size_t bhb = (size_t)(b * NHEAD + h);
          size_t idx = (mode == 1) ? (bhb * SEQ + s) * HDIM + hd
                                   : (bhb * HDIM + hd) * SEQ + s;
          C[idx] = (_Float16)acc[ai][t][i];
        }
  }
}

// --------------------------- flash attention -------------------------------
// One wave per (bh, 16-row q tile): stream K/V in 32-column blocks with
// online softmax.  K-tiles for the next block and V-tiles for the current
// block are loaded before the softmax VALU work so vmem overlaps it.
// Qp,Kp: [BH,S,HD] f16;  VpT: [BH,HD,S] f16;  Oh: f16 [B*S, D].
__global__ __launch_bounds__(128) void flash_attn(
    const _Float16* __restrict__ Qp, const _Float16* __restrict__ Kp,
    const _Float16* __restrict__ VpT, const int* __restrict__ valid_length,
    _Float16* __restrict__ Oh) {
  __shared__ _Float16 plds[4][16 * 32];

  const int lane = threadIdx.x & 31;
  const int w    = threadIdx.x >> 5;
  const int wf   = blockIdx.x * 4 + w;     // 8192 waves
  const int bh   = wf >> 7;                // 64 head-batches
  const int qt   = wf & 127;               // 128 q tiles
  const int r    = lane & 15;
  const int hi   = lane >> 4;
  // faithful to reference: valid_flat[bh] = valid_length[bh % B]
  const int valid = valid_length[bh & (BATCH - 1)];

  const _Float16* Qb = Qp  + (size_t)bh * SEQ * HDIM;
  const _Float16* Kb = Kp  + (size_t)bh * SEQ * HDIM;
  const _Float16* Vb = VpT + (size_t)bh * SEQ * HDIM;

  v16h aq0 = load_a_tile(Qb + (size_t)(qt * 16 + r) * HDIM + hi * 8);
  v16h aq1 = load_a_tile(Qb + (size_t)(qt * 16 + r) * HDIM + 32 + hi * 8);

  v8f o[4] = {};
  float mrow[8], lrow[8];
#pragma unroll
  for (int i = 0; i < 8; ++i) { mrow[i] = -3.0e38f; lrow[i] = 0.0f; }

  _Float16* P = plds[w];

  // prologue: K-block 0 B-tiles
  const _Float16* kp0 = Kb + (size_t)r * HDIM + hi * 8;
  v16h kb00 = load_b_tile(kp0);
  v16h kb01 = load_b_tile(kp0 + 32);
  v16h kb10 = load_b_tile(kp0 + 16 * HDIM);
  v16h kb11 = load_b_tile(kp0 + 16 * HDIM + 32);

  for (int kb = 0; kb < SEQ; kb += 32) {
    // ---- S = (Q K^T) * 1/sqrt(HD), 16x32 block -----------------------------
    v8f s0 = {}, s1 = {};
    s0 = wmma_f16(aq0, kb00, s0);
    s0 = wmma_f16(aq1, kb01, s0);
    s1 = wmma_f16(aq0, kb10, s1);
    s1 = wmma_f16(aq1, kb11, s1);

    // prefetch next K-block B-tiles (wraps on last iter; harmless)
    const int kn = (kb + 32) & (SEQ - 1);
    const _Float16* knp = Kb + (size_t)(kn + r) * HDIM + hi * 8;
    v16h nk00 = load_b_tile(knp);
    v16h nk01 = load_b_tile(knp + 32);
    v16h nk10 = load_b_tile(knp + 16 * HDIM);
    v16h nk11 = load_b_tile(knp + 16 * HDIM + 32);
    // prefetch V B-tiles for the current block
    const _Float16* vp = Vb + (size_t)r * SEQ + kb + hi * 8;
    v16h vb0 = load_b_tile(vp);
    v16h vb1 = load_b_tile(vp + 16 * SEQ);
    v16h vb2 = load_b_tile(vp + 32 * SEQ);
    v16h vb3 = load_b_tile(vp + 48 * SEQ);

    // ---- scale + length mask (column uniform per lane) ---------------------
    const bool msk0 = (kb + r)      >= valid;
    const bool msk1 = (kb + 16 + r) >= valid;
#pragma unroll
    for (int i = 0; i < 8; ++i) {
      s0[i] = msk0 ? -1.0e6f : s0[i] * 0.125f;
      s1[i] = msk1 ? -1.0e6f : s1[i] * 0.125f;
    }

    // ---- online softmax (rows live across 16 lanes of each half-wave) ------
#pragma unroll
    for (int i = 0; i < 8; ++i) {
      float bm    = rmax16(fmaxf(s0[i], s1[i]));
      float mn    = fmaxf(mrow[i], bm);
      float alpha = __expf(mrow[i] - mn);
      float p0    = __expf(s0[i] - mn);
      float p1    = __expf(s1[i] - mn);
      lrow[i] = lrow[i] * alpha + rsum16(p0 + p1);
      mrow[i] = mn;
      o[0][i] *= alpha; o[1][i] *= alpha; o[2][i] *= alpha; o[3][i] *= alpha;
      // C-layout -> row-major 16x32 f16 P in LDS
      P[(i + 8 * hi) * 32 + r]      = (_Float16)p0;
      P[(i + 8 * hi) * 32 + 16 + r] = (_Float16)p1;
    }
    asm volatile("s_wait_dscnt 0" ::: "memory");  // same-wave LDS RAW

    // ---- O += P @ V  (P reloaded in A layout from LDS) ---------------------
    v16h ap = load_a_tile(P + r * 32 + hi * 8);
    o[0] = wmma_f16(ap, vb0, o[0]);
    o[1] = wmma_f16(ap, vb1, o[1]);
    o[2] = wmma_f16(ap, vb2, o[2]);
    o[3] = wmma_f16(ap, vb3, o[3]);

    kb00 = nk00; kb01 = nk01; kb10 = nk10; kb11 = nk11;
  }

  // ---- normalize + merge heads: Oh[b*S + q, h*64 + col] --------------------
  const int b = bh >> 4, h = bh & 15;
#pragma unroll
  for (int t = 0; t < 4; ++t)
#pragma unroll
    for (int i = 0; i < 8; ++i) {
      int q   = qt * 16 + 8 * hi + i;
      int col = h * HDIM + t * 16 + r;
      Oh[(size_t)(b * SEQ + q) * DMODEL + col] = (_Float16)(o[t][i] / lrow[i]);
    }
}

// ------------------------------ launcher -----------------------------------
extern "C" void kernel_launch(void* const* d_in, const int* in_sizes, int n_in,
                              void* d_out, int out_size, void* d_ws,
                              size_t ws_size, hipStream_t stream) {
  (void)in_sizes; (void)n_in; (void)out_size; (void)ws_size;
  const float* q     = (const float*)d_in[0];
  const float* k     = (const float*)d_in[1];
  const float* v     = (const float*)d_in[2];
  const int*   vlen  = (const int*)d_in[3];
  const float* Wq    = (const float*)d_in[4];
  const float* Wk    = (const float*)d_in[5];
  const float* Wv    = (const float*)d_in[6];
  const float* Wo    = (const float*)d_in[7];

  const size_t NX = (size_t)MROWS * DMODEL;   // 8388608
  const size_t NW = (size_t)DMODEL * DMODEL;  // 1048576

  _Float16* ws   = (_Float16*)d_ws;
  _Float16* qh   = ws;
  _Float16* kh   = qh + NX;
  _Float16* vh   = kh + NX;
  _Float16* WqT  = vh + NX;
  _Float16* WkT  = WqT + NW;
  _Float16* WvT  = WkT + NW;
  _Float16* WoT  = WvT + NW;
  _Float16* Qp   = WoT + NW;        // [BH,S,HD]
  _Float16* Kp   = Qp + NX;         // [BH,S,HD]
  _Float16* VpT  = Kp + NX;         // [BH,HD,S]
  _Float16* attn = VpT + NX;        // [B*S, D]

  // 1) f32 -> f16 conversions (+ LDS-tiled weight transposes)
  cvt_h4<<<(int)(NX / 1024), 256, 0, stream>>>(q, qh);
  cvt_h4<<<(int)(NX / 1024), 256, 0, stream>>>(k, kh);
  cvt_h4<<<(int)(NX / 1024), 256, 0, stream>>>(v, vh);
  cvt_hT<<<1024, 256, 0, stream>>>(Wq, WqT);
  cvt_hT<<<1024, 256, 0, stream>>>(Wk, WkT);
  cvt_hT<<<1024, 256, 0, stream>>>(Wv, WvT);
  cvt_hT<<<1024, 256, 0, stream>>>(Wo, WoT);

  // 2) Q/K/V projections into split-head layouts (WMMA GEMMs, 32x64 tiles)
  const int gemm_blocks = (MROWS / 32) * (DMODEL / 64) / 4;  // 1024
  gemm_wmma<<<gemm_blocks, 128, 0, stream>>>(qh, WqT, (void*)Qp, 1);
  gemm_wmma<<<gemm_blocks, 128, 0, stream>>>(kh, WkT, (void*)Kp, 1);
  gemm_wmma<<<gemm_blocks, 128, 0, stream>>>(vh, WvT, (void*)VpT, 2);

  // 3) flash attention: 64 head-batches x 128 q-tiles = 8192 waves
  flash_attn<<<(BH * (SEQ / 16)) / 4, 128, 0, stream>>>(Qp, Kp, VpT, vlen, attn);

  // 4) final projection -> f32 d_out
  gemm_wmma<<<gemm_blocks, 128, 0, stream>>>(attn, WoT, d_out, 0);
}
